// GraphCAD_82781199663583
// MI455X (gfx1250) — matrix operand order repaired and checked by reading, not compiled, not run on hardware
//
#include <hip/hip_runtime.h>
#include <hip/hip_bf16.h>
#include <math.h>
#include <stdint.h>

typedef __attribute__((ext_vector_type(16))) _Float16 v16h;
typedef __attribute__((ext_vector_type(8)))  _Float16 v8h;
typedef __attribute__((ext_vector_type(2)))  _Float16 v2h;
typedef __attribute__((ext_vector_type(8)))  float    v8f;
typedef __attribute__((ext_vector_type(4)))  unsigned v4u;
typedef __attribute__((ext_vector_type(8)))  int      v8i;
typedef __attribute__((ext_vector_type(4)))  int      v4i;

#define DD    64
#define BSZ   16
#define NPG   4096
#define NNODE (BSZ*NPG)      // 65536
#define NEDGE (NNODE*16)     // 1048576
#define TEMPR 0.6f
#define BNEPS 1e-5f

#define EST 200   // LDS row stride (halves) for K=192 tiles
#define NST 72    // LDS row stride (halves) for K=64 tiles

// ---- Tensor Data Mover: 1-D tile (contiguous bytes) global -> LDS ----
// D# per CDNA5 ISA ch.8: group0 {count=1, lds_addr, global_addr, type=2},
// group1 {mask=0, data_size=8B, tensor_dim0=n8, tensor_dim1=1, tile_dim0=n8,
// tensor_dim0_stride=n8}. bytes must be multiple of 8; tile_dim0 fits 16 bits.
__device__ __forceinline__ void tdm_load_to_lds(void* ldsDst, const void* gSrc, unsigned bytes) {
  unsigned ldsOff = (unsigned)(uintptr_t)(__attribute__((address_space(3))) char*)ldsDst;
  unsigned long long ga = (unsigned long long)(uintptr_t)gSrc;
  unsigned n8 = bytes >> 3;
  v4u g0;
  g0[0] = 1u;                                   // count=1 (valid descriptor)
  g0[1] = ldsOff;                               // lds_addr
  g0[2] = (unsigned)ga;                         // global_addr[31:0]
  g0[3] = (unsigned)((ga >> 32) & 0x01FFFFFFu) | (2u << 30);  // ga[56:32] | type=2
  v8i g1;
  g1[0] = 0x00030000;                           // data_size=3 (8B), wg_mask=0
  g1[1] = (int)((n8 & 0xFFFFu) << 16);          // tensor_dim0[15:0] @ bits 63:48
  g1[2] = (int)((n8 >> 16) | (1u << 16));       // tensor_dim0 hi | tensor_dim1=1
  g1[3] = (int)((n8 & 0xFFFFu) << 16);          // tile_dim0 @ bits 127:112
  g1[4] = 0;                                    // tile_dim1=0, tile_dim2=0
  g1[5] = (int)n8;                              // tensor_dim0_stride lo
  g1[6] = 0;
  g1[7] = 0;
  v4i gz = {0, 0, 0, 0};
#if __clang_major__ >= 23
  v8i gz8 = {0, 0, 0, 0, 0, 0, 0, 0};
  __builtin_amdgcn_tensor_load_to_lds(g0, g1, gz, gz, gz8, 0);
#else
  __builtin_amdgcn_tensor_load_to_lds(g0, g1, gz, gz, 0);
#endif
}

// ---- WMMA fragment loaders (wave32 layouts per CDNA5 ISA 7.12.2) ----
__device__ __forceinline__ v16h frag_a(const _Float16* base, int row, int stride, int kb, int lane) {
  const _Float16* p = base + row * stride + kb + ((lane & 16) ? 8 : 0);
  v8h lo = *(const v8h*)p;
  v8h hi = *(const v8h*)(p + 16);
  return __builtin_shufflevector(lo, hi, 0,1,2,3,4,5,6,7,8,9,10,11,12,13,14,15);
}
__device__ __forceinline__ v16h frag_b(const _Float16* base, int col, int stride, int kb, int lane) {
  const _Float16* p = base + col * stride + kb + ((lane & 16) ? 8 : 0);
  v8h lo = *(const v8h*)p;
  v8h hi = *(const v8h*)(p + 16);
  return __builtin_shufflevector(lo, hi, 0,1,2,3,4,5,6,7,8,9,10,11,12,13,14,15);
}

// ---- weight image prep: f32 [K][64] -> f16 transposed [64][stride] (zero pad) ----
__global__ void transpose_f16_kernel(const float* __restrict__ W, _Float16* __restrict__ img,
                                     int K, int stride) {
  int i = blockIdx.x * 256 + threadIdx.x;
  if (i < 64 * stride) {
    int n = i / stride, k = i - n * stride;
    img[i] = (k < K) ? (_Float16)W[k * 64 + n] : (_Float16)0.f;
  }
}

// ---- per-graph mean (initial centroid) ----
__global__ __launch_bounds__(256) void centroid_init_kernel(const float* __restrict__ x,
                                                            float* __restrict__ cent) {
  __shared__ float red[256];
  int g = blockIdx.x, tid = threadIdx.x;
  int fe = tid & 63, grp = tid >> 6;
  const float* xg = x + (size_t)g * NPG * DD;
  float acc = 0.f;
  for (int n = grp; n < NPG; n += 4) acc += xg[(size_t)n * DD + fe];
  red[tid] = acc;
  __syncthreads();
  if (tid < 64) cent[g * DD + tid] =
      (red[tid] + red[64 + tid] + red[128 + tid] + red[192 + tid]) * (1.f / NPG);
}

// ---- edge scorer: TDM weight tile + fused gather + 2x [16,192]x[192,64] WMMA ----
__global__ __launch_bounds__(128) void edge_kernel(
    const float* __restrict__ x, const float* __restrict__ cent,
    const int* __restrict__ ei, const float* __restrict__ ew_in,
    const float* __restrict__ un, const _Float16* __restrict__ wimg,
    const float* __restrict__ b1, const float* __restrict__ W2,
    const float* __restrict__ b2, const float* __restrict__ alpha,
    float* __restrict__ ew_out, float* __restrict__ pre_out) {
  __shared__ alignas(16) _Float16 sW[64 * EST];       // 25600 B (TDM-loaded)
  __shared__ alignas(16) _Float16 sA[4][32 * EST];    // 51200 B per-wave staging
  __shared__ alignas(16) float    sC[BSZ * DD];       // 4096 B

  int tid = threadIdx.x;
  if (tid < 32) tdm_load_to_lds(sW, wimg, 64 * EST * 2);
  for (int i = tid; i < BSZ * DD; i += 128) sC[i] = cent[i];
  __builtin_amdgcn_s_wait_tensorcnt(0);
  __syncthreads();

  int wave = tid >> 5, lane = tid & 31, nl = lane & 15, hi16 = lane >> 4;
  _Float16* stg = &sA[wave][0];
  const int* srcA = ei;
  const int* dstA = ei + NEDGE;
  int ebase = blockIdx.x * 128 + wave * 32;
  int f = lane * 2;

  for (int m = 0; m < 32; ++m) {
    int e = ebase + m;
    int s = srcA[e], d = dstA[e];
    float2 xs = *(const float2*)(x + (size_t)s * DD + f);
    float2 xd = *(const float2*)(x + (size_t)d * DD + f);
    const float* cd = sC + (d >> 12) * DD + f;
    const float* cs = sC + (s >> 12) * DD + f;
    _Float16* row = stg + m * EST;
    v2h t0; t0[0] = (_Float16)fabsf(xd.x - xs.x); t0[1] = (_Float16)fabsf(xd.y - xs.y);
    v2h t1; t1[0] = (_Float16)(xd.x - cd[0]);     t1[1] = (_Float16)(xd.y - cd[1]);
    v2h t2; t2[0] = (_Float16)(xs.x - cs[0]);     t2[1] = (_Float16)(xs.y - cs[1]);
    *(v2h*)(row + f)       = t0;
    *(v2h*)(row + 64 + f)  = t1;
    *(v2h*)(row + 128 + f) = t2;
  }
  // wave-private staging: same-wave DS ops are in-order, no barrier needed

  v8f acc0[4] = {}, acc1[4] = {};
  float b1v[4], w2v[4];
  #pragma unroll
  for (int nt = 0; nt < 4; ++nt) { b1v[nt] = b1[nt * 16 + nl]; w2v[nt] = W2[nt * 16 + nl]; }

  #pragma unroll
  for (int kc = 0; kc < 6; ++kc) {
    v16h A0 = frag_a(stg, nl, EST, kc * 32, lane);
    v16h A1 = frag_a(stg + 16 * EST, nl, EST, kc * 32, lane);
    #pragma unroll
    for (int nt = 0; nt < 4; ++nt) {
      v16h B = frag_b(sW, nt * 16 + nl, EST, kc * 32, lane);  // B reused by both M-tiles
      acc0[nt] = __builtin_amdgcn_wmma_f32_16x16x32_f16(false, A0, false, B,
                                                        (short)0, acc0[nt], false, false);
      acc1[nt] = __builtin_amdgcn_wmma_f32_16x16x32_f16(false, A1, false, B,
                                                        (short)0, acc1[nt], false, false);
    }
  }

  float p0[8], p1[8];
  #pragma unroll
  for (int r = 0; r < 8; ++r) { p0[r] = 0.f; p1[r] = 0.f; }
  #pragma unroll
  for (int nt = 0; nt < 4; ++nt) {
    #pragma unroll
    for (int r = 0; r < 8; ++r) {
      p0[r] = fmaf(fmaxf(acc0[nt][r] + b1v[nt], 0.f), w2v[nt], p0[r]);
      p1[r] = fmaf(fmaxf(acc1[nt][r] + b1v[nt], 0.f), w2v[nt], p1[r]);
    }
  }
  #pragma unroll
  for (int r = 0; r < 8; ++r) {
    p0[r] += __shfl_xor(p0[r], 1, 32); p0[r] += __shfl_xor(p0[r], 2, 32);
    p0[r] += __shfl_xor(p0[r], 4, 32); p0[r] += __shfl_xor(p0[r], 8, 32);
    p1[r] += __shfl_xor(p1[r], 1, 32); p1[r] += __shfl_xor(p1[r], 2, 32);
    p1[r] += __shfl_xor(p1[r], 4, 32); p1[r] += __shfl_xor(p1[r], 8, 32);
  }
  float bb2 = b2[0], al = alpha[0];
  #pragma unroll
  for (int r = 0; r < 8; ++r) {
    if (nl == r) {
      #pragma unroll
      for (int t = 0; t < 2; ++t) {
        int e = ebase + t * 16 + r + hi16 * 8;
        float score = (t ? p1[r] : p0[r]) + bb2;
        float pre = 1.f / (1.f + __expf(-score));
        float u = un[e];
        float g = __logf(u) - log1pf(-u);
        float soft = 1.f / (1.f + __expf(-(score + g) * (1.f / TEMPR)));
        float samp = rintf(soft);                     // straight-through forward
        float ewv = ew_in[e];
        ew_out[e] = al * (samp * ewv) + (1.f - al) * (samp * pre);
        pre_out[e] = pre;
      }
    }
  }
}

// ---- GIN scatter: aggr[dst] += ew * x[src] ----
__global__ __launch_bounds__(256) void scatter_kernel(const int* __restrict__ ei,
                                                      const float* __restrict__ ew,
                                                      const float* __restrict__ x,
                                                      float* __restrict__ aggr) {
  int gid = blockIdx.x * 256 + threadIdx.x;
  int e = gid >> 5;
  int f = (gid & 31) * 2;
  float w = ew[e];
  if (w != 0.f) {
    int s = ei[e], d = ei[NEDGE + e];
    float2 xv = *(const float2*)(x + (size_t)s * DD + f);
    atomicAdd(aggr + (size_t)d * DD + f,     w * xv.x);
    atomicAdd(aggr + (size_t)d * DD + f + 1, w * xv.y);
  }
}

__global__ void copy_kernel(const float* __restrict__ src, float* __restrict__ dst, int n) {
  int i = blockIdx.x * 256 + threadIdx.x;
  if (i < n) dst[i] = src[i];
}

__global__ void stats_zero_kernel(float* stats) {
  if (threadIdx.x < 128) stats[threadIdx.x] = 0.f;
}

// ---- node GEMM: xh = relu((x+aggr)@W + b), in-place, + BN stats ----
__global__ __launch_bounds__(128) void node_mm_kernel(
    const float* __restrict__ ain, const _Float16* __restrict__ wimg,
    const float* __restrict__ b, float* __restrict__ xh_out,
    float* __restrict__ stats) {
  __shared__ alignas(16) _Float16 sW[64 * NST];
  __shared__ alignas(16) _Float16 sA[4][16 * NST];
  __shared__ float sstat[128];
  int tid = threadIdx.x;
  if (tid < 32) tdm_load_to_lds(sW, wimg, 64 * NST * 2);
  if (tid < 128) sstat[tid] = 0.f;
  __builtin_amdgcn_s_wait_tensorcnt(0);
  __syncthreads();

  int wave = tid >> 5, lane = tid & 31, nl = lane & 15, hi16 = lane >> 4;
  _Float16* stg = &sA[wave][0];
  int rbase = blockIdx.x * 64 + wave * 16;
  int f = lane * 2;
  for (int m = 0; m < 16; ++m) {
    float2 v = *(const float2*)(ain + (size_t)(rbase + m) * DD + f);
    v2h t; t[0] = (_Float16)v.x; t[1] = (_Float16)v.y;
    *(v2h*)(stg + m * NST + f) = t;
  }
  v8f acc[4] = {};
  float bv[4];
  #pragma unroll
  for (int nt = 0; nt < 4; ++nt) bv[nt] = b[nt * 16 + nl];
  #pragma unroll
  for (int kc = 0; kc < 2; ++kc) {
    v16h A = frag_a(stg, nl, NST, kc * 32, lane);
    #pragma unroll
    for (int nt = 0; nt < 4; ++nt) {
      v16h B = frag_b(sW, nt * 16 + nl, NST, kc * 32, lane);
      acc[nt] = __builtin_amdgcn_wmma_f32_16x16x32_f16(false, A, false, B,
                                                       (short)0, acc[nt], false, false);
    }
  }
  float psum[4] = {0.f, 0.f, 0.f, 0.f}, psq[4] = {0.f, 0.f, 0.f, 0.f};
  #pragma unroll
  for (int nt = 0; nt < 4; ++nt) {
    #pragma unroll
    for (int r = 0; r < 8; ++r) {
      float v = fmaxf(acc[nt][r] + bv[nt], 0.f);
      int m = r + hi16 * 8;
      xh_out[(size_t)(rbase + m) * DD + nt * 16 + nl] = v;
      psum[nt] += v;
      psq[nt] = fmaf(v, v, psq[nt]);
    }
  }
  #pragma unroll
  for (int nt = 0; nt < 4; ++nt) {
    psum[nt] += __shfl_xor(psum[nt], 16, 32);
    psq[nt]  += __shfl_xor(psq[nt], 16, 32);
  }
  if (lane < 16) {
    #pragma unroll
    for (int nt = 0; nt < 4; ++nt) {
      atomicAdd(&sstat[nt * 16 + lane], psum[nt]);
      atomicAdd(&sstat[64 + nt * 16 + lane], psq[nt]);
    }
  }
  __syncthreads();
  if (tid < 128) atomicAdd(&stats[tid], sstat[tid]);
}

__global__ void stats_final_kernel(float* stats, const float* __restrict__ gamma,
                                   const float* __restrict__ beta) {
  int t = threadIdx.x;
  if (t < 64) {
    float mu = stats[t] * (1.f / NNODE);
    float var = stats[64 + t] * (1.f / NNODE) - mu * mu;
    float sc = gamma[t] * rsqrtf(var + BNEPS);
    stats[128 + t] = sc;
    stats[192 + t] = beta[t] - mu * sc;
  }
}

__global__ void bn_apply_kernel(const float* __restrict__ xh, const float* __restrict__ stats,
                                float* __restrict__ xout) {
  int i = blockIdx.x * 256 + threadIdx.x;
  int f = i & 63;
  xout[i] = xh[i] * stats[128 + f] + stats[192 + f];
}

// ---- attention pooling per graph ----
__global__ __launch_bounds__(256) void system_kernel(const float* __restrict__ x,
                                                     float* __restrict__ cent,
                                                     float* __restrict__ tbuf) {
  __shared__ float sc[64];
  __shared__ float red[256];
  int g = blockIdx.x, tid = threadIdx.x;
  if (tid < 64) sc[tid] = cent[g * DD + tid];
  __syncthreads();
  const float* xg = x + (size_t)g * NPG * DD;
  float lmax = -3.4e38f;
  for (int n = tid; n < NPG; n += 256) {
    const float* xr = xg + (size_t)n * DD;
    float t = 0.f;
    for (int k = 0; k < 64; ++k) t = fmaf(xr[k], sc[k], t);
    tbuf[g * NPG + n] = t;
    lmax = fmaxf(lmax, t);
  }
  red[tid] = lmax; __syncthreads();
  for (int s = 128; s > 0; s >>= 1) { if (tid < s) red[tid] = fmaxf(red[tid], red[tid + s]); __syncthreads(); }
  float bmax = red[0]; __syncthreads();
  float lsum = 0.f;
  for (int n = tid; n < NPG; n += 256) lsum += __expf(tbuf[g * NPG + n] - bmax);
  red[tid] = lsum; __syncthreads();
  for (int s = 128; s > 0; s >>= 1) { if (tid < s) red[tid] += red[tid + s]; __syncthreads(); }
  float bsum = red[0]; __syncthreads();
  int fe = tid & 63, grp = tid >> 6;
  float accf = 0.f;
  for (int n = grp; n < NPG; n += 4) {
    float w = __expf(tbuf[g * NPG + n] - bmax);
    accf = fmaf(w, xg[(size_t)n * DD + fe], accf);
  }
  red[tid] = accf; __syncthreads();
  if (tid < 64) cent[g * DD + tid] =
      (red[tid] + red[64 + tid] + red[128 + tid] + red[192 + tid]) / bsum;
}

// ---- head: two chained 64x64 WMMA GEMMs (x rows + centroid tile) ----
__global__ __launch_bounds__(128) void head_kernel(
    const float* __restrict__ x, const float* __restrict__ cent,
    const _Float16* __restrict__ wimg,   // W1^T then W2^T, contiguous
    const float* __restrict__ b1, const float* __restrict__ b2,
    float* __restrict__ xl, float* __restrict__ cl) {
  __shared__ alignas(16) _Float16 sW12[2 * 64 * NST];
  __shared__ alignas(16) _Float16 sA[4][16 * NST];
  int tid = threadIdx.x;
  if (tid < 32) tdm_load_to_lds(sW12, wimg, 2 * 64 * NST * 2);
  __builtin_amdgcn_s_wait_tensorcnt(0);
  __syncthreads();
  int wave = tid >> 5, lane = tid & 31, nl = lane & 15, hi16 = lane >> 4;
  int tile = blockIdx.x * 4 + wave;
  if (tile > 4096) return;   // no barriers after this point
  const float* rin; float* rout;
  if (tile == 4096) { rin = cent; rout = cl; }
  else { rin = x + (size_t)tile * 16 * DD; rout = xl + (size_t)tile * 16 * DD; }
  _Float16* stg = &sA[wave][0];
  int f = lane * 2;
  for (int m = 0; m < 16; ++m) {
    float2 v = *(const float2*)(rin + (size_t)m * DD + f);
    v2h t; t[0] = (_Float16)v.x; t[1] = (_Float16)v.y;
    *(v2h*)(stg + m * NST + f) = t;
  }
  v8f acc[4] = {};
  float b1v[4], b2v[4];
  #pragma unroll
  for (int nt = 0; nt < 4; ++nt) { b1v[nt] = b1[nt * 16 + nl]; b2v[nt] = b2[nt * 16 + nl]; }
  #pragma unroll
  for (int kc = 0; kc < 2; ++kc) {
    v16h A = frag_a(stg, nl, NST, kc * 32, lane);
    #pragma unroll
    for (int nt = 0; nt < 4; ++nt) {
      v16h B = frag_b(sW12, nt * 16 + nl, NST, kc * 32, lane);
      acc[nt] = __builtin_amdgcn_wmma_f32_16x16x32_f16(false, A, false, B,
                                                       (short)0, acc[nt], false, false);
    }
  }
  #pragma unroll
  for (int nt = 0; nt < 4; ++nt) {
    #pragma unroll
    for (int r = 0; r < 8; ++r) {
      int m = r + hi16 * 8;
      stg[m * NST + nt * 16 + nl] = (_Float16)fmaxf(acc[nt][r] + b1v[nt], 0.f);
    }
  }
  v8f acc2[4] = {};
  #pragma unroll
  for (int kc = 0; kc < 2; ++kc) {
    v16h A = frag_a(stg, nl, NST, kc * 32, lane);
    #pragma unroll
    for (int nt = 0; nt < 4; ++nt) {
      v16h B = frag_b(sW12 + 64 * NST, nt * 16 + nl, NST, kc * 32, lane);
      acc2[nt] = __builtin_amdgcn_wmma_f32_16x16x32_f16(false, A, false, B,
                                                        (short)0, acc2[nt], false, false);
    }
  }
  #pragma unroll
  for (int nt = 0; nt < 4; ++nt) {
    #pragma unroll
    for (int r = 0; r < 8; ++r) {
      int m = r + hi16 * 8;
      rout[(size_t)m * DD + nt * 16 + nl] = acc2[nt][r] + b2v[nt];
    }
  }
}

__global__ void final_copy_kernel(const float* __restrict__ xcur, const float* __restrict__ cent,
                                  float* __restrict__ out_x, float* __restrict__ out_c) {
  int i = blockIdx.x * 256 + threadIdx.x;
  out_x[i] = xcur[i];
  if (i < BSZ * DD) out_c[i] = cent[i];
}

extern "C" void kernel_launch(void* const* d_in, const int* in_sizes, int n_in,
                              void* d_out, int out_size, void* d_ws, size_t ws_size,
                              hipStream_t stream) {
  (void)in_sizes; (void)n_in; (void)out_size; (void)ws_size;
  const float* x_in  = (const float*)d_in[0];
  const int*   ei    = (const int*)d_in[1];
  const float* ew_in = (const float*)d_in[2];
  const float* un    = (const float*)d_in[5];
  const float* epW1  = (const float*)d_in[6];
  const float* epb1  = (const float*)d_in[7];
  const float* epW2  = (const float*)d_in[8];
  const float* epb2  = (const float*)d_in[9];
  const float* alpha = (const float*)d_in[10];
  const float* ginW  = (const float*)d_in[11];
  const float* ginb  = (const float*)d_in[12];
  const float* bng   = (const float*)d_in[13];
  const float* bnb   = (const float*)d_in[14];
  const float* hW1   = (const float*)d_in[15];
  const float* hb1   = (const float*)d_in[16];
  const float* hW2   = (const float*)d_in[17];
  const float* hb2   = (const float*)d_in[18];

  float* out  = (float*)d_out;
  float* o_x  = out;
  float* o_ew = out + (size_t)NNODE * DD;
  float* o_c  = o_ew + NEDGE;
  float* o_xl = o_c + BSZ * DD;
  float* o_cl = o_xl + (size_t)NNODE * DD;
  float* o_pa = o_cl + BSZ * DD;

  float* ws    = (float*)d_ws;
  float* xcur  = ws;                              // N*D
  float* aggr  = xcur + (size_t)NNODE * DD;       // N*D (doubles as xh)
  float* cent  = aggr + (size_t)NNODE * DD;       // BS*D
  float* stats = cent + BSZ * DD;                 // 256
  float* tbuf  = stats + 256;                     // N
  _Float16* wimg_edge = (_Float16*)(tbuf + NNODE);        // 64*EST halves
  _Float16* wimg_gin  = wimg_edge + 64 * EST;             // 2 x 64*NST
  _Float16* wimg_head = wimg_gin + 2 * 64 * NST;          // 2 x 64*NST (W1,W2)

  // weight images (once; weights are loop-invariant)
  transpose_f16_kernel<<<(64 * EST + 255) / 256, 256, 0, stream>>>(epW1, wimg_edge, 192, EST);
  transpose_f16_kernel<<<(64 * NST + 255) / 256, 256, 0, stream>>>(ginW, wimg_gin, 64, NST);
  transpose_f16_kernel<<<(64 * NST + 255) / 256, 256, 0, stream>>>(ginW + DD * DD,
                                                                   wimg_gin + 64 * NST, 64, NST);
  transpose_f16_kernel<<<(64 * NST + 255) / 256, 256, 0, stream>>>(hW1, wimg_head, 64, NST);
  transpose_f16_kernel<<<(64 * NST + 255) / 256, 256, 0, stream>>>(hW2, wimg_head + 64 * NST, 64, NST);

  centroid_init_kernel<<<BSZ, 256, 0, stream>>>(x_in, cent);

  const float* xs = x_in;
  for (int l = 0; l < 2; ++l) {
    const float* ewsrc = (l == 0) ? ew_in : o_ew;
    edge_kernel<<<NEDGE / 128, 128, 0, stream>>>(xs, cent, ei, ewsrc,
        un + (size_t)l * NEDGE, wimg_edge, epb1, epW2, epb2, alpha, o_ew, o_pa);
    copy_kernel<<<(NNODE * DD) / 256, 256, 0, stream>>>(xs, aggr, NNODE * DD);
    stats_zero_kernel<<<1, 128, 0, stream>>>(stats);
    scatter_kernel<<<(NEDGE / 8), 256, 0, stream>>>(ei, o_ew, xs, aggr);
    node_mm_kernel<<<NNODE / 64, 128, 0, stream>>>(aggr, wimg_gin + (size_t)l * 64 * NST,
                                                   ginb + l * DD, aggr, stats);
    stats_final_kernel<<<1, 64, 0, stream>>>(stats, bng + l * DD, bnb + l * DD);
    bn_apply_kernel<<<(NNODE * DD) / 256, 256, 0, stream>>>(aggr, stats, xcur);
    system_kernel<<<BSZ, 256, 0, stream>>>(xcur, cent, tbuf);
    xs = xcur;
  }
  head_kernel<<<(4096 / 4) + 1, 128, 0, stream>>>(xcur, cent, wimg_head, hb1, hb2, o_xl, o_cl);
  final_copy_kernel<<<(NNODE * DD) / 256, 256, 0, stream>>>(xcur, cent, o_x, o_c);
}
